// StackLSTM_87222195848024
// MI455X (gfx1250) — matrix operand order repaired and checked by reading, not compiled, not run on hardware
//
#include <hip/hip_runtime.h>

// ---------------- problem constants ----------------
#define T_STEPS 1024
#define BATCH   128
#define HID     128
#define NLAYER  2
#define STACK   1024

#define NWG     8             // cooperating workgroups (cluster-sized)
#define UPW     (HID / NWG)   // 16 hidden units owned per WG
#define NROWS   (4 * UPW)     // 64 gate rows per layer owned per WG
#define LDW     136           // padded LDS row stride (bf16 elems) for weights
#define THREADS 256           // 8 wave32 waves

typedef __attribute__((ext_vector_type(16))) __bf16 v16bf;
typedef __attribute__((ext_vector_type(8)))  float  v8f;

// ---------------- workspace layout (bytes) ----------------
// All WMMA A-operand streams are bf16: recurrent h stacks, layer exchange, and a
// one-time bf16 copy of `inputs`. c stacks stay fp32 (additive cell path).
#define STACK_ELEMS ((size_t)(STACK + 1) * BATCH * HID)
#define X_ELEMS     ((size_t)T_STEPS * BATCH * HID)
#define OFF_BAR   0
#define OFF_HTMP  1024                                        // bf16 (B,H) layer exchange
#define OFF_HS0   ((size_t)OFF_HTMP + (size_t)BATCH * HID * 2)
#define OFF_HS1   (OFF_HS0 + STACK_ELEMS * 2)                 // bf16 mirror of contents
#define OFF_CS0   (OFF_HS1 + STACK_ELEMS * 2)
#define OFF_CS1   (OFF_CS0 + STACK_ELEMS * 4)
#define OFF_XBF   (OFF_CS1 + STACK_ELEMS * 4)                 // bf16 copy of inputs

__device__ __forceinline__ float sigmoidf_(float x) { return 1.0f / (1.0f + __expf(-x)); }
__device__ __forceinline__ float tanhf_(float x)    { return 2.0f / (1.0f + __expf(-2.0f * x)) - 1.0f; }

// A-fragment (16x32 bf16) for V_WMMA_F32_16X16X32_BF16 from a per-lane bf16 row.
// ISA layout: lanes0-15 M=lane&15; VGPR r<4: K=k0+hi*8+2r, r>=4: K=k0+16+hi*8+2(r-4).
__device__ __forceinline__ v16bf frag_a_row(const __bf16* __restrict__ rowbase,
                                            int k0, int hi) {
  v16bf a;
#pragma unroll
  for (int r = 0; r < 8; ++r) {
    int bk = k0 + ((r & 4) ? 16 : 0) + hi * 8 + (r & 3) * 2;
    a[2 * r]     = rowbase[bk];
    a[2 * r + 1] = rowbase[bk + 1];
  }
  return a;
}

// B-fragment (32x16 bf16): lane holds column n = lane&15; elem i<8: K = hi*8 + i, i>=8: K = 16 + hi*8 + (i-8).
// B[k][n] == W[n_row][k]; weights row-major in LDS with stride LDW.
__device__ __forceinline__ v16bf frag_b_from_lds(const __bf16* __restrict__ wbase,
                                                 int nrow0, int k0, int n, int hi) {
  const __bf16* rp = wbase + (size_t)(nrow0 + n) * LDW + k0 + hi * 8;
  v16bf b;
#pragma unroll
  for (int i = 0; i < 16; ++i) b[i] = rp[i + ((i & 8) ? 8 : 0)];
  return b;
}

// gates(Btile, 64) = X*Wih^T + H*Whh^T accumulated in 4 f32 16x16 tiles (i,f,g,o)
__device__ __forceinline__ void gemm_gates(const __bf16* __restrict__ xrow,
                                           const __bf16* __restrict__ hrow,
                                           const __bf16* __restrict__ wih,
                                           const __bf16* __restrict__ whh,
                                           int nn, int hi, v8f acc[4]) {
#pragma unroll
  for (int k0 = 0; k0 < HID; k0 += 32) {
    v16bf aX = frag_a_row(xrow, k0, hi);
    v16bf aH = frag_a_row(hrow, k0, hi);
#pragma unroll
    for (int nt = 0; nt < 4; ++nt) {
      v16bf bI = frag_b_from_lds(wih, nt * 16, k0, nn, hi);
      v16bf bH = frag_b_from_lds(whh, nt * 16, k0, nn, hi);
      acc[nt] = __builtin_amdgcn_wmma_f32_16x16x32_bf16(false, aX, false, bI,
                                                        (short)0, acc[nt], false, false);
      acc[nt] = __builtin_amdgcn_wmma_f32_16x16x32_bf16(false, aH, false, bH,
                                                        (short)0, acc[nt], false, false);
    }
  }
}

// grid-wide barrier: global atomic sense barrier + CDNA5 cluster barrier
// (s_barrier_signal/wait -3; NOP when ClusterID==0).
__device__ __forceinline__ void grid_sync(unsigned* bar, int nblk) {
  __syncthreads();
  if (threadIdx.x == 0) {
    __threadfence();
    volatile unsigned* sense = bar + 1;
    unsigned s = *sense;
    if (atomicAdd(bar, 1u) == (unsigned)(nblk - 1)) {
      *bar = 0u;
      __threadfence();
      atomicAdd((unsigned*)(bar + 1), 1u);
    } else {
      while (*sense == s) { __builtin_amdgcn_s_sleep(8); }
    }
    __threadfence();
  }
  __builtin_amdgcn_s_cluster_barrier();
  __syncthreads();
}

__global__ __launch_bounds__(THREADS)
void stacklstm_kernel(const int*   __restrict__ ops,      // (T,B)
                      const float* __restrict__ w_ih,     // (L,4H,H)
                      const float* __restrict__ w_hh,     // (L,4H,H)
                      const float* __restrict__ b_ih,     // (L,4H)
                      const float* __restrict__ b_hh,     // (L,4H)
                      float* __restrict__ out,            // contents (S+1,B,H) ++ pos(B)
                      unsigned char* __restrict__ ws) {
  // Persistent weight slices (bf16) + fused biases + stack positions in LDS.
  __shared__ __bf16 sW[NLAYER][2][NROWS][LDW];   // [layer][ih/hh][gate-row][k]
  __shared__ float  sBias[NLAYER][NROWS];
  __shared__ int    sPos[BATCH];

  const int g    = blockIdx.x;      // owns hidden units [g*UPW, (g+1)*UPW)
  const int tid  = threadIdx.x;
  const int lane = tid & 31;
  const int wv   = tid >> 5;        // wave id == batch M-tile
  const int hi   = lane >> 4;
  const int nn   = lane & 15;

  unsigned*     bar      = (unsigned*)(ws + OFF_BAR);
  __bf16*       htmp     = (__bf16*)(ws + OFF_HTMP);
  __bf16*       hS0      = (__bf16*)(ws + OFF_HS0);
  __bf16*       hS1      = (__bf16*)(ws + OFF_HS1);   // bf16 mirror of `contents`
  float*        cS0      = (float*)(ws + OFF_CS0);
  float*        cS1      = (float*)(ws + OFF_CS1);
  const __bf16* xbf      = (const __bf16*)(ws + OFF_XBF);
  float*        contents = out;

  // ---- one-time: convert owned weight rows to bf16 in LDS ----
  for (int idx = tid; idx < NLAYER * 2 * NROWS; idx += THREADS) {
    int l = idx / (2 * NROWS);
    int m = (idx / NROWS) & 1;
    int r = idx % NROWS;
    int q = r / UPW, u = r % UPW;
    int grow = q * HID + g * UPW + u;   // global gate row
    const float* src = (m == 0 ? w_ih : w_hh) + ((size_t)l * 4 * HID + grow) * HID;
    for (int k = 0; k < HID; ++k) sW[l][m][r][k] = (__bf16)src[k];
  }
  for (int idx = tid; idx < NLAYER * NROWS; idx += THREADS) {
    int l = idx / NROWS, r = idx % NROWS;
    int q = r / UPW, u = r % UPW;
    int grow = q * HID + g * UPW + u;
    sBias[l][r] = b_ih[(size_t)l * 4 * HID + grow] + b_hh[(size_t)l * 4 * HID + grow];
  }
  if (tid < BATCH) sPos[tid] = 0;
  grid_sync(bar, NWG);   // also orders after the init kernel's stores

  const int unit = g * UPW + nn;   // elementwise column handled by this thread

  for (int t = 0; t < T_STEPS; ++t) {
    const int mb = wv * 16 + nn;   // per-lane A-matrix batch row

    // Prefetch next timestep's x rows (global_prefetch_b8).
    if (t + 1 < T_STEPS)
      __builtin_prefetch(xbf + ((size_t)(t + 1) * BATCH + mb) * HID, 0, 3);

    // ---------------- layer 0 ----------------
    {
      const int posb = sPos[mb];
      const __bf16* xrow = xbf + ((size_t)t * BATCH + mb) * HID;
      const __bf16* hrow = hS0 + ((size_t)posb * BATCH + mb) * HID;
      v8f acc[4] = {};
      gemm_gates(xrow, hrow, &sW[0][0][0][0], &sW[0][1][0][0], nn, hi, acc);

      const float bI = sBias[0][0 * UPW + nn], bF = sBias[0][1 * UPW + nn];
      const float bG = sBias[0][2 * UPW + nn], bO = sBias[0][3 * UPW + nn];
#pragma unroll
      for (int j = 0; j < 8; ++j) {
        int   bb = wv * 16 + hi * 8 + j;
        int   pb = sPos[bb];
        float iv = sigmoidf_(acc[0][j] + bI);
        float fv = sigmoidf_(acc[1][j] + bF);
        float gv = tanhf_(acc[2][j] + bG);
        float ov = sigmoidf_(acc[3][j] + bO);
        float cp = cS0[((size_t)pb * BATCH + bb) * HID + unit];
        float cn = fv * cp + iv * gv;
        float hn = ov * tanhf_(cn);
        size_t wr = ((size_t)(pb + 1) * BATCH + bb) * HID + unit;
        cS0[wr] = cn;
        __bf16 hb = (__bf16)hn;
        hS0[wr] = hb;                          // bf16: next-step A operand
        htmp[(size_t)bb * HID + unit] = hb;    // bf16: layer-1 input exchange
      }
    }
    grid_sync(bar, NWG);   // htmp complete across all WGs

    // ---------------- layer 1 ----------------
    {
      const int posb = sPos[mb];
      const __bf16* xrow = htmp + (size_t)mb * HID;
      const __bf16* hrow = hS1 + ((size_t)posb * BATCH + mb) * HID;
      v8f acc[4] = {};
      gemm_gates(xrow, hrow, &sW[1][0][0][0], &sW[1][1][0][0], nn, hi, acc);

      const float bI = sBias[1][0 * UPW + nn], bF = sBias[1][1 * UPW + nn];
      const float bG = sBias[1][2 * UPW + nn], bO = sBias[1][3 * UPW + nn];
#pragma unroll
      for (int j = 0; j < 8; ++j) {
        int   bb = wv * 16 + hi * 8 + j;
        int   pb = sPos[bb];
        float iv = sigmoidf_(acc[0][j] + bI);
        float fv = sigmoidf_(acc[1][j] + bF);
        float gv = tanhf_(acc[2][j] + bG);
        float ov = sigmoidf_(acc[3][j] + bO);
        float cp = cS1[((size_t)pb * BATCH + bb) * HID + unit];
        float cn = fv * cp + iv * gv;
        float hn = ov * tanhf_(cn);
        size_t wr = ((size_t)(pb + 1) * BATCH + bb) * HID + unit;
        cS1[wr] = cn;
        contents[wr] = hn;          // fp32 output (h_stack layer L-1)
        hS1[wr] = (__bf16)hn;       // bf16 mirror for next-step gather
      }
    }

    // ---- advance stack positions (identical copy per WG) ----
    __syncthreads();
    if (tid < BATCH) sPos[tid] += ops[(size_t)t * BATCH + tid];
    grid_sync(bar, NWG);   // stack writes + pos visible before next gather
  }

  if (g == 0 && tid < BATCH) {
    ((int*)out)[(size_t)(STACK + 1) * BATCH * HID + tid] = sPos[tid];
  }
}

// Fully parallel setup pass (off the serial critical path):
//  * zero output contents (unwritten stack slots must read as 0)
//  * zero stack entry 0 of all state buffers, htmp, barrier words
//  * convert the entire `inputs` tensor to bf16 once (x is consumed once;
//    this removes all per-step f32->bf16 conversion from the recurrent loop)
__global__ void stacklstm_init_kernel(const float* __restrict__ inputs,
                                      float* __restrict__ out,
                                      unsigned char* __restrict__ ws) {
  size_t i = (size_t)blockIdx.x * blockDim.x + threadIdx.x;
  size_t stride = (size_t)gridDim.x * blockDim.x;

  const size_t n_out = (size_t)(STACK + 1) * BATCH * HID + BATCH;
  for (size_t k = i; k < n_out; k += stride) out[k] = 0.f;

  __bf16* xbf = (__bf16*)(ws + OFF_XBF);
  for (size_t k = i; k < X_ELEMS; k += stride) xbf[k] = (__bf16)inputs[k];

  __bf16* hS0  = (__bf16*)(ws + OFF_HS0);
  __bf16* hS1  = (__bf16*)(ws + OFF_HS1);
  float*  cS0  = (float*)(ws + OFF_CS0);
  float*  cS1  = (float*)(ws + OFF_CS1);
  __bf16* htmp = (__bf16*)(ws + OFF_HTMP);
  for (size_t k = i; k < (size_t)BATCH * HID; k += stride) {
    hS0[k] = (__bf16)0.f; hS1[k] = (__bf16)0.f;
    cS0[k] = 0.f; cS1[k] = 0.f;
    htmp[k] = (__bf16)0.f;
  }
  if (i < 2) ((unsigned*)(ws + OFF_BAR))[i] = 0u;
}

extern "C" void kernel_launch(void* const* d_in, const int* in_sizes, int n_in,
                              void* d_out, int out_size, void* d_ws, size_t ws_size,
                              hipStream_t stream) {
  (void)in_sizes; (void)n_in; (void)out_size; (void)ws_size;
  const float* inputs = (const float*)d_in[0];
  const int*   ops    = (const int*)d_in[1];
  const float* w_ih   = (const float*)d_in[2];
  const float* w_hh   = (const float*)d_in[3];
  const float* b_ih   = (const float*)d_in[4];
  const float* b_hh   = (const float*)d_in[5];
  // d_in[6] = stack_size scalar (compile-time constant here)

  stacklstm_init_kernel<<<1024, 256, 0, stream>>>(inputs, (float*)d_out,
                                                  (unsigned char*)d_ws);
  stacklstm_kernel<<<NWG, THREADS, 0, stream>>>(ops, w_ih, w_hh, b_ih, b_hh,
                                                (float*)d_out, (unsigned char*)d_ws);
}
